// MultiHeadAttention_62156766708361
// MI455X (gfx1250) — matrix operand order repaired and checked by reading, not compiled, not run on hardware
//
#include <hip/hip_runtime.h>
#include <hip/hip_bf16.h>
#include <math.h>

// ---------------------------------------------------------------------------
// MHA forward for MI455X (gfx1250): bf16 WMMA everywhere, flash softmax,
// async global->LDS staging + software pipelining.
// ---------------------------------------------------------------------------

typedef __attribute__((ext_vector_type(16))) __bf16 v16bf;
typedef __attribute__((ext_vector_type(8)))  float  v8f;
typedef __attribute__((ext_vector_type(4)))  int    v4i;

union FragB16 { v16bf v; uint4 q[2]; };

static constexpr int kB    = 4;
static constexpr int kSeq  = 2048;
static constexpr int kEmb  = 768;
static constexpr int kHeads= 8;
static constexpr int kHd   = 96;     // head dim
static constexpr int kNqkv = 2304;

__device__ __forceinline__ unsigned short f2bf(float f) {
  unsigned int u = __float_as_uint(f);
  u += 0x7FFFu + ((u >> 16) & 1u);   // round-to-nearest-even
  return (unsigned short)(u >> 16);
}
__device__ __forceinline__ unsigned int pack2(float a, float b) {
  return (unsigned int)f2bf(a) | ((unsigned int)f2bf(b) << 16);
}

// ---- async global->LDS copy (16B per lane), ASYNCcnt-tracked ---------------
__device__ __forceinline__ void async_copy16(const unsigned short* g,
                                             unsigned short* l) {
#if __has_builtin(__builtin_amdgcn_global_load_async_to_lds_b128)
  __builtin_amdgcn_global_load_async_to_lds_b128(
      (__attribute__((address_space(1))) v4i*)g,
      (__attribute__((address_space(3))) v4i*)l, 0, 0);
#else
  const unsigned loff = (unsigned)(unsigned long long)l;   // LDS = addr[31:0]
  asm volatile("global_load_async_to_lds_b128 %0, %1, off"
               :: "v"(loff), "v"((unsigned long long)g) : "memory");
#endif
}
__device__ __forceinline__ void wait_async0() {
#if __has_builtin(__builtin_amdgcn_s_wait_asynccnt)
  __builtin_amdgcn_s_wait_asynccnt(0);
#else
  asm volatile("s_wait_asynccnt 0" ::: "memory");
#endif
}

// ---------------------------------------------------------------------------
// Generic 128x128x32-tiled bf16 GEMM, 8 waves, each wave = 32x64 (2x4 WMMA).
// Register-prefetch software pipeline: next tile's global loads issue while
// the current tile's WMMAs run.
// MODE 0: A fp32 (= x), B fp32 (= Wqkv); epilogue scatters to Q/K/V bf16,
//         Q pre-scaled by 1/sqrt(emb).
// MODE 1: A bf16 (= attn out), B fp32 (= Wproj); epilogue bias + fp32 store.
// ---------------------------------------------------------------------------
template<int MODE>
__global__ __launch_bounds__(256)
void gemm_bf16_wmma(const void* __restrict__ Ap, const float* __restrict__ Bw,
                    const float* __restrict__ bias, int M, int N, int Kd,
                    unsigned short* __restrict__ qb, unsigned short* __restrict__ kbuf,
                    unsigned short* __restrict__ vbuf, float* __restrict__ outF)
{
  __shared__ __align__(16) unsigned short As[128][40];   // [row][k], pitch 80B
  __shared__ __align__(16) unsigned short Bs[128][40];   // [col][k], pitch 80B

  const int m0 = blockIdx.y * 128;
  const int n0 = blockIdx.x * 128;
  const int t  = threadIdx.x;
  const int w  = t >> 5, lane = t & 31;
  const int wm = w >> 1, wn = w & 1;          // 4x2 wave grid
  const int lr = lane & 15, hisel = lane >> 4;

  // staging coordinates (fixed per thread)
  const int arow = t >> 1, aseg = (t & 1) * 16;        // A: 16 elems/thread
  const int bcol = t & 127, bkg = (t >> 7) & 1;        // B: 16 elems/thread

  v8f acc[2][4];
  #pragma unroll
  for (int mt = 0; mt < 2; ++mt)
    #pragma unroll
    for (int nt = 0; nt < 4; ++nt)
      #pragma unroll
      for (int r = 0; r < 8; ++r) acc[mt][nt][r] = 0.f;

  // prefetch registers
  float4 fa[4]; uint4 ua[2]; float fb[16];

  auto loadTiles = [&](int k0) {
    if constexpr (MODE == 0) {
      const float* src = (const float*)Ap + (size_t)(m0 + arow) * Kd + k0 + aseg;
      fa[0] = ((const float4*)src)[0];
      fa[1] = ((const float4*)src)[1];
      fa[2] = ((const float4*)src)[2];
      fa[3] = ((const float4*)src)[3];
    } else {
      const unsigned short* src =
          (const unsigned short*)Ap + (size_t)(m0 + arow) * Kd + k0 + aseg;
      ua[0] = ((const uint4*)src)[0];
      ua[1] = ((const uint4*)src)[1];
    }
    #pragma unroll
    for (int j = 0; j < 16; ++j)
      fb[j] = Bw[(size_t)(k0 + bkg * 16 + j) * N + n0 + bcol];
  };

  auto storeTiles = [&]() {
    if constexpr (MODE == 0) {
      uint4 s0, s1;
      s0.x = pack2(fa[0].x, fa[0].y); s0.y = pack2(fa[0].z, fa[0].w);
      s0.z = pack2(fa[1].x, fa[1].y); s0.w = pack2(fa[1].z, fa[1].w);
      s1.x = pack2(fa[2].x, fa[2].y); s1.y = pack2(fa[2].z, fa[2].w);
      s1.z = pack2(fa[3].x, fa[3].y); s1.w = pack2(fa[3].z, fa[3].w);
      *(uint4*)&As[arow][aseg]     = s0;
      *(uint4*)&As[arow][aseg + 8] = s1;
    } else {
      *(uint4*)&As[arow][aseg]     = ua[0];
      *(uint4*)&As[arow][aseg + 8] = ua[1];
    }
    uint4 s0, s1;
    s0.x = pack2(fb[0], fb[1]);   s0.y = pack2(fb[2], fb[3]);
    s0.z = pack2(fb[4], fb[5]);   s0.w = pack2(fb[6], fb[7]);
    s1.x = pack2(fb[8], fb[9]);   s1.y = pack2(fb[10], fb[11]);
    s1.z = pack2(fb[12], fb[13]); s1.w = pack2(fb[14], fb[15]);
    *(uint4*)&Bs[bcol][bkg * 16]     = s0;
    *(uint4*)&Bs[bcol][bkg * 16 + 8] = s1;
  };

  loadTiles(0);
  for (int k0 = 0; k0 < Kd; k0 += 32) {
    storeTiles();
    __syncthreads();
    if (k0 + 32 < Kd) loadTiles(k0 + 32);   // overlap with WMMAs below

    FragB16 a[2], b[4];
    #pragma unroll
    for (int mt = 0; mt < 2; ++mt) {
      const int row = wm * 32 + mt * 16 + lr;
      a[mt].q[0] = *(const uint4*)&As[row][hisel * 8];        // K = hi*8 + 0..7
      a[mt].q[1] = *(const uint4*)&As[row][hisel * 8 + 16];   // K = 16+hi*8 ..
    }
    #pragma unroll
    for (int nt = 0; nt < 4; ++nt) {
      const int col = wn * 64 + nt * 16 + lr;
      b[nt].q[0] = *(const uint4*)&Bs[col][hisel * 8];        // K = hi*8 + 0..7
      b[nt].q[1] = *(const uint4*)&Bs[col][hisel * 8 + 16];   // K = 16+hi*8 ..
    }
    #pragma unroll
    for (int mt = 0; mt < 2; ++mt)
      #pragma unroll
      for (int nt = 0; nt < 4; ++nt)
        acc[mt][nt] = __builtin_amdgcn_wmma_f32_16x16x32_bf16(
            false, a[mt].v, false, b[nt].v, (short)0, acc[mt][nt], false, false);
    __syncthreads();
  }

  // ---- epilogue ----
  #pragma unroll
  for (int mt = 0; mt < 2; ++mt) {
    const int rowb = m0 + wm * 32 + mt * 16 + hisel * 8;
    #pragma unroll
    for (int nt = 0; nt < 4; ++nt) {
      const int col = n0 + wn * 64 + nt * 16 + lr;
      const float bv = bias[col];
      #pragma unroll
      for (int r = 0; r < 8; ++r) {
        const float val = acc[mt][nt][r] + bv;
        const int grow = rowb + r;
        if constexpr (MODE == 0) {
          // col = (h*96 + d)*3 + qkv  (qkv innermost per reference reshape)
          const int qkv = col % 3;
          const int hd  = col / 3;
          const int h   = hd / kHd;
          const int d   = hd - h * kHd;
          const int bi  = grow >> 11;           // / 2048
          const int np  = grow & 2047;
          const size_t idx =
              ((size_t)(bi * kHeads + h) * kSeq + np) * kHd + d;
          if (qkv == 0)      qb[idx]   = f2bf(val * 0.03608439182f); // 1/sqrt(768)
          else if (qkv == 1) kbuf[idx] = f2bf(val);
          else               vbuf[idx] = f2bf(val);
        } else {
          outF[(size_t)grow * N + col] = val;
        }
      }
    }
  }
}

// ---------------------------------------------------------------------------
// Flash attention: one block per (b*h, 128 query rows). 8 waves x 16 q-rows.
// Q fragments resident in VGPRs. K tiles double-buffered in LDS via async
// global->LDS copies (issued for tile kt+1 during tile kt's WMMAs); V tiles
// pipelined through registers (2-byte transpose, not async-expressible).
// ---------------------------------------------------------------------------
__global__ __launch_bounds__(256)
void attn_wmma(const unsigned short* __restrict__ qb,
               const unsigned short* __restrict__ kb,
               const unsigned short* __restrict__ vb,
               unsigned short* __restrict__ ao)
{
  __shared__ __align__(16) unsigned short Ks[2][64][104];  // [key][d] pitch 208B
  __shared__ __align__(16) unsigned short Vt[96][72];      // [d][key] pitch 144B
  __shared__ __align__(16) unsigned short Ps[8][16][72];   // wave-private P

  const int bh = blockIdx.y;
  const int q0 = blockIdx.x * 128;
  const int t = threadIdx.x, w = t >> 5, lane = t & 31;
  const int lr = lane & 15, hisel = lane >> 4;

  // Q fragments for this wave's 16 rows: 3 chunks of K(d)=32
  FragB16 qf[3];
  {
    const size_t rowbase = ((size_t)bh * kSeq + (q0 + w * 16 + lr)) * kHd;
    #pragma unroll
    for (int c = 0; c < 3; ++c) {
      const unsigned short* p = qb + rowbase + c * 32 + hisel * 8;
      qf[c].q[0] = ((const uint4*)p)[0];
      qf[c].q[1] = *(const uint4*)(p + 16);
    }
  }

  float mrow[8], lrow[8];
  #pragma unroll
  for (int r = 0; r < 8; ++r) { mrow[r] = -3.0e38f; lrow[r] = 0.f; }
  v8f o[6];
  #pragma unroll
  for (int dt = 0; dt < 6; ++dt)
    #pragma unroll
    for (int r = 0; r < 8; ++r) o[dt][r] = 0.f;

  const size_t kvbh = (size_t)bh * kSeq * kHd;
  constexpr int NT = kSeq / 64;

  auto stageK = [&](int kt, int buf) {     // async: 3 x 16B per thread
    const unsigned short* gk = kb + kvbh + (size_t)kt * 64 * kHd;
    #pragma unroll
    for (int i = 0; i < 3; ++i) {
      const int e0 = (t + i * 256) * 8;
      const int key = e0 / kHd, d = e0 - key * kHd;   // 8 | 96 -> in-row
      async_copy16(gk + e0, &Ks[buf][key][d]);
    }
  };

  unsigned short vr[24];                   // V pipeline registers
  auto loadV = [&](int kt) {
    const unsigned short* gv = vb + kvbh + (size_t)kt * 64 * kHd;
    #pragma unroll
    for (int i = 0; i < 24; ++i) vr[i] = gv[t + i * 256];
  };
  auto storeV = [&]() {                    // transpose into Vt[d][key]
    #pragma unroll
    for (int i = 0; i < 24; ++i) {
      const int id = t + i * 256;
      const int key = id / kHd, d = id - key * kHd;
      Vt[d][key] = vr[i];
    }
  };

  // prologue: stage tile 0
  stageK(0, 0);
  loadV(0);
  storeV();

  for (int kt = 0; kt < NT; ++kt) {
    wait_async0();          // this wave's K-tile asyncs done
    __syncthreads();        // whole block's staging visible; prior reads done
    const int p = kt & 1;
    if (kt + 1 < NT) {
      stageK(kt + 1, p ^ 1);   // overlaps with WMMAs below
      loadV(kt + 1);           // global loads complete during compute
    }

    // ---- S = Q * K^T : 4 key sub-tiles x 3 K-chunks = 12 WMMA ----
    v8f s[4];
    #pragma unroll
    for (int j = 0; j < 4; ++j) {
      #pragma unroll
      for (int r = 0; r < 8; ++r) s[j][r] = 0.f;
      const int keyl = j * 16 + lr;
      #pragma unroll
      for (int c = 0; c < 3; ++c) {
        FragB16 bk;
        bk.q[0] = *(const uint4*)&Ks[p][keyl][c * 32 + hisel * 8];
        bk.q[1] = *(const uint4*)&Ks[p][keyl][c * 32 + hisel * 8 + 16];
        s[j] = __builtin_amdgcn_wmma_f32_16x16x32_bf16(
            false, qf[c].v, false, bk.v, (short)0, s[j], false, false);
      }
    }

    // ---- online softmax (row stats per-lane, C-layout row r+hi*8) ----
    float rmax[8], rsum[8], alpha[8];
    #pragma unroll
    for (int r = 0; r < 8; ++r) rmax[r] = -3.0e38f;
    #pragma unroll
    for (int j = 0; j < 4; ++j)
      #pragma unroll
      for (int r = 0; r < 8; ++r) rmax[r] = fmaxf(rmax[r], s[j][r]);
    #pragma unroll
    for (int r = 0; r < 8; ++r) {
      #pragma unroll
      for (int off = 1; off < 16; off <<= 1)
        rmax[r] = fmaxf(rmax[r], __shfl_xor(rmax[r], off, 32));
      const float mn = fmaxf(mrow[r], rmax[r]);
      alpha[r] = __expf(mrow[r] - mn);
      mrow[r] = mn;
      rsum[r] = 0.f;
    }
    #pragma unroll
    for (int j = 0; j < 4; ++j)
      #pragma unroll
      for (int r = 0; r < 8; ++r) {
        const float pv = __expf(s[j][r] - mrow[r]);
        s[j][r] = pv;
        rsum[r] += pv;
      }
    #pragma unroll
    for (int r = 0; r < 8; ++r) {
      #pragma unroll
      for (int off = 1; off < 16; off <<= 1)
        rsum[r] += __shfl_xor(rsum[r], off, 32);
      lrow[r] = lrow[r] * alpha[r] + rsum[r];
    }
    #pragma unroll
    for (int dt = 0; dt < 6; ++dt)
      #pragma unroll
      for (int r = 0; r < 8; ++r) o[dt][r] *= alpha[r];

    // ---- P: C-layout -> LDS -> A-layout (wave-private, DS in-order) ----
    #pragma unroll
    for (int j = 0; j < 4; ++j)
      #pragma unroll
      for (int r = 0; r < 8; ++r)
        Ps[w][r + hisel * 8][j * 16 + lr] = f2bf(s[j][r]);

    FragB16 pf[2];
    #pragma unroll
    for (int c = 0; c < 2; ++c) {
      pf[c].q[0] = *(const uint4*)&Ps[w][lr][c * 32 + hisel * 8];
      pf[c].q[1] = *(const uint4*)&Ps[w][lr][c * 32 + hisel * 8 + 16];
    }

    // ---- O += P * V : 6 d-tiles x 2 key-chunks = 12 WMMA ----
    #pragma unroll
    for (int dt = 0; dt < 6; ++dt) {
      const int dl = dt * 16 + lr;
      #pragma unroll
      for (int c = 0; c < 2; ++c) {
        FragB16 bvf;
        bvf.q[0] = *(const uint4*)&Vt[dl][c * 32 + hisel * 8];
        bvf.q[1] = *(const uint4*)&Vt[dl][c * 32 + hisel * 8 + 16];
        o[dt] = __builtin_amdgcn_wmma_f32_16x16x32_bf16(
            false, pf[c].v, false, bvf.v, (short)0, o[dt], false, false);
      }
    }
    __syncthreads();          // all waves done reading Vt
    if (kt + 1 < NT) storeV(); // write next V tile (visible after next barrier)
  }

  // ---- epilogue: normalize + store bf16 [b, n, h*96+d] ----
  const int bi = bh / kHeads, h = bh - bi * kHeads;
  const int rowb = q0 + w * 16 + hisel * 8;
  #pragma unroll
  for (int dt = 0; dt < 6; ++dt) {
    const int d = dt * 16 + lr;
    #pragma unroll
    for (int r = 0; r < 8; ++r) {
      const float val = o[dt][r] / lrow[r];
      ao[((size_t)(bi * kSeq + rowb + r)) * kEmb + h * kHd + d] = f2bf(val);
    }
  }
}

// ---------------------------------------------------------------------------
extern "C" void kernel_launch(void* const* d_in, const int* in_sizes, int n_in,
                              void* d_out, int out_size, void* d_ws, size_t ws_size,
                              hipStream_t stream) {
  const float* x     = (const float*)d_in[0];
  const float* Wqkv  = (const float*)d_in[1];
  const float* bqkv  = (const float*)d_in[2];
  const float* Wproj = (const float*)d_in[3];
  const float* bproj = (const float*)d_in[4];
  float* out = (float*)d_out;

  const size_t ELEMS = (size_t)kB * kHeads * kSeq * kHd;   // 6,291,456
  unsigned short* qb = (unsigned short*)d_ws;
  unsigned short* kb = qb + ELEMS;
  unsigned short* vb = kb + ELEMS;
  unsigned short* ao = vb + ELEMS;                          // 8192x768 bf16

  const dim3 blk(256);
  gemm_bf16_wmma<0><<<dim3(kNqkv / 128, (kB * kSeq) / 128), blk, 0, stream>>>(
      x, Wqkv, bqkv, kB * kSeq, kNqkv, kEmb, qb, kb, vb, nullptr);
  attn_wmma<<<dim3(kSeq / 128, kB * kHeads), blk, 0, stream>>>(qb, kb, vb, ao);
  gemm_bf16_wmma<1><<<dim3(kEmb / 128, (kB * kSeq) / 128), blk, 0, stream>>>(
      ao, Wproj, bproj, kB * kSeq, kEmb, kEmb, nullptr, nullptr, nullptr, out);
}
